// LSTMusic_35579509080317
// MI455X (gfx1250) — matrix operand order repaired and checked by reading, not compiled, not run on hardware
//
#include <hip/hip_runtime.h>
#include <hip/hip_bf16.h>

// ---------------------------------------------------------------------------
// LSTMusic on MI455X (gfx1250, wave32, WMMA)
//
// Persistent-workgroup design: grid = B/16 = 32 blocks, 1024 threads (32
// waves). Each block owns a 16-row batch tile for the full 256-step
// recurrence. c stays in registers (v8f per wave), h in LDS (bf16).
// All GEMMs use v_wmma_f32_16x16x32_bf16 with bf16 weights prepared by small
// conversion/transpose kernels into the workspace.
// ---------------------------------------------------------------------------

#define BB 512
#define HIST 64
#define SEQN 256
#define GENN 192
#define DD 2
#define HH 512
#define H2 1024
#define G4H 2048
#define NPITCH 129
#define NOUT 130
#define NPAD 144

typedef __bf16 bhalf;
typedef bhalf v16bf __attribute__((ext_vector_type(16)));
typedef float v8f __attribute__((ext_vector_type(8)));
typedef unsigned int u32x4 __attribute__((ext_vector_type(4)));

union Frag {
    v16bf v;
    u32x4 q[2];
};

__device__ __forceinline__ unsigned short f2bf(float f) {
    unsigned int u = __float_as_uint(f);
    u += 0x7fffu + ((u >> 16) & 1u);   // round-to-nearest-even
    return (unsigned short)(u >> 16);
}
__device__ __forceinline__ float sigm(float x) { return 1.0f / (1.0f + __expf(-x)); }
__device__ __forceinline__ float tanh_fast(float x) { return 1.0f - 2.0f / (1.0f + __expf(2.0f * x)); }
__device__ __forceinline__ v8f v8zero() {
    v8f v;
#pragma unroll
    for (int i = 0; i < 8; ++i) v[i] = 0.f;
    return v;
}

// -------------------------- prep kernels -----------------------------------
__global__ void k_cvt_bf16(const float* __restrict__ src, unsigned short* __restrict__ dst, int n) {
    int i = blockIdx.x * blockDim.x + threadIdx.x;
    if (i < n) dst[i] = f2bf(src[i]);
}

// dst[n*cols_out + k] = (n < src_rows_valid) ? src[k*src_cols + n] : 0   (bf16)
__global__ void k_transpose_bf16(const float* __restrict__ src, unsigned short* __restrict__ dst,
                                 int rows_out, int cols_out, int src_cols, int src_rows_valid) {
    int i = blockIdx.x * blockDim.x + threadIdx.x;
    if (i >= rows_out * cols_out) return;
    int n = i / cols_out;
    int k = i - n * cols_out;
    float v = (n < src_rows_valid) ? src[(size_t)k * src_cols + n] : 0.f;
    dst[i] = f2bf(v);
}

__global__ void k_bias_sum(const float* __restrict__ a, const float* __restrict__ b,
                           float* __restrict__ c, int n) {
    int i = blockIdx.x * blockDim.x + threadIdx.x;
    if (i < n) c[i] = a[i] + b[i];
}

// -------------------------- main persistent kernel -------------------------
__global__ __launch_bounds__(1024, 1)
void lstmusic_persistent(const float* __restrict__ hist_x,
                         const float* __restrict__ noise_u,
                         const float* __restrict__ noise_g,
                         const float* __restrict__ W_ih,
                         const unsigned short* __restrict__ Whh,   // (2048,512) bf16 row-major
                         const unsigned short* __restrict__ W1T,   // (1024,512) bf16
                         const unsigned short* __restrict__ W2T,   // (512,1024) bf16
                         const unsigned short* __restrict__ W3T,   // (144,512) bf16 (zero-padded)
                         const float* __restrict__ bsum,           // b_ih + b_hh (2048)
                         const float* __restrict__ b1v,
                         const float* __restrict__ b2v,
                         const float* __restrict__ b3v,
                         float* __restrict__ out) {
    constexpr int HP = HH + 8;     // bf16 pitch, multiple of 8 -> 16B-aligned rows
    constexpr int Z1P = H2 + 8;
    constexpr int Z3P = 160;

    extern __shared__ char smem[];
    unsigned short* h_lds = (unsigned short*)smem;      // 16*HP
    unsigned short* z1_lds = h_lds + 16 * HP;           // 16*Z1P
    unsigned short* z2_lds = z1_lds + 16 * Z1P;         // 16*HP
    float* z3_lds = (float*)(z2_lds + 16 * HP);         // 16*Z3P
    float* x_lds = z3_lds + 16 * Z3P;                   // 16*2

    const int tid = threadIdx.x;
    const int lane = tid & 31;
    const int wid = tid >> 5;            // 0..31
    const int nsub = lane & 15;          // N column (B/C/D) or M row (A)
    const bool hi = lane >= 16;
    const int aOff = hi ? 8 : 0;         // A-fragment K sub-offset
    const int bOff = hi ? 8 : 0;         // B-fragment K sub-offset
    const int mBase = hi ? 8 : 0;        // C/D row base
    const int rowBase = blockIdx.x * 16; // global batch rows of this block
    const int jt = wid * 16;             // hidden-tile base for this wave

    // zero h; copy history slice of the output
    for (int i = tid; i < 16 * HP; i += blockDim.x) h_lds[i] = 0;
    for (int i = tid; i < 16 * HIST * DD; i += blockDim.x) {
        int r = i / (HIST * DD);
        int rem = i - r * (HIST * DD);
        out[((size_t)(rowBase + r) * SEQN) * DD + rem] =
            hist_x[((size_t)(rowBase + r) * HIST) * DD + rem];
    }

    v8f creg = v8zero();   // cell state tile: rows mBase..mBase+7, col jt+nsub
    __syncthreads();

    for (int step = 0; step < SEQN; ++step) {
        if (step < HIST && tid < 32) {   // load x_t for conditioning phase
            int r = tid >> 1, d = tid & 1;
            x_lds[tid] = hist_x[((size_t)(rowBase + r) * HIST + step) * DD + d];
        }

        // ---- gates = h @ Whh^T  (4 x 16x16 tiles per wave, K = 512) ----
        v8f gi = v8zero(), gf = v8zero(), gg = v8zero(), go = v8zero();
        for (int kk = 0; kk < HH; kk += 32) {
            Frag a;
            a.q[0] = *(const u32x4*)&h_lds[nsub * HP + kk + aOff];
            a.q[1] = *(const u32x4*)&h_lds[nsub * HP + kk + 16 + aOff];
            const unsigned short* w0 = Whh + (size_t)(0 * HH + jt + nsub) * HH + kk + bOff;
            const unsigned short* w1 = Whh + (size_t)(1 * HH + jt + nsub) * HH + kk + bOff;
            const unsigned short* w2 = Whh + (size_t)(2 * HH + jt + nsub) * HH + kk + bOff;
            const unsigned short* w3 = Whh + (size_t)(3 * HH + jt + nsub) * HH + kk + bOff;
            Frag b0, b1f, b2f, b3f;
            b0.q[0] = *(const u32x4*)w0;  b0.q[1] = *(const u32x4*)(w0 + 16);
            b1f.q[0] = *(const u32x4*)w1; b1f.q[1] = *(const u32x4*)(w1 + 16);
            b2f.q[0] = *(const u32x4*)w2; b2f.q[1] = *(const u32x4*)(w2 + 16);
            b3f.q[0] = *(const u32x4*)w3; b3f.q[1] = *(const u32x4*)(w3 + 16);
            gi = __builtin_amdgcn_wmma_f32_16x16x32_bf16(false, a.v, false, b0.v,  (short)0, gi, false, false);
            gf = __builtin_amdgcn_wmma_f32_16x16x32_bf16(false, a.v, false, b1f.v, (short)0, gf, false, false);
            gg = __builtin_amdgcn_wmma_f32_16x16x32_bf16(false, a.v, false, b2f.v, (short)0, gg, false, false);
            go = __builtin_amdgcn_wmma_f32_16x16x32_bf16(false, a.v, false, b3f.v, (short)0, go, false, false);
        }

        const int nH = jt + nsub;  // hidden unit owned by this lane's column
        float wi0 = W_ih[(0 * HH + nH) * DD + 0], wi1 = W_ih[(0 * HH + nH) * DD + 1];
        float wf0 = W_ih[(1 * HH + nH) * DD + 0], wf1 = W_ih[(1 * HH + nH) * DD + 1];
        float wg0 = W_ih[(2 * HH + nH) * DD + 0], wg1 = W_ih[(2 * HH + nH) * DD + 1];
        float wo0 = W_ih[(3 * HH + nH) * DD + 0], wo1 = W_ih[(3 * HH + nH) * DD + 1];
        float bi = bsum[0 * HH + nH], bfv = bsum[1 * HH + nH];
        float bg = bsum[2 * HH + nH], bo = bsum[3 * HH + nH];

        __syncthreads();  // all h reads done; x_lds visible

        float hnew[8];
#pragma unroll
        for (int v = 0; v < 8; ++v) {
            int m = mBase + v;
            float x0 = x_lds[m * 2 + 0];
            float x1 = x_lds[m * 2 + 1];
            float vi = gi[v] + wi0 * x0 + wi1 * x1 + bi;
            float vf = gf[v] + wf0 * x0 + wf1 * x1 + bfv;
            float vg = gg[v] + wg0 * x0 + wg1 * x1 + bg;
            float vo = go[v] + wo0 * x0 + wo1 * x1 + bo;
            float ct = sigm(vf) * creg[v] + sigm(vi) * tanh_fast(vg);
            creg[v] = ct;
            hnew[v] = sigm(vo) * tanh_fast(ct);
        }
#pragma unroll
        for (int v = 0; v < 8; ++v)
            h_lds[(mBase + v) * HP + nH] = f2bf(hnew[v]);
        __syncthreads();

        if (step < HIST) continue;
        const int s = step - HIST;

        // ---- z1 = tanh(h @ W1 + b1) : 16x1024, 2 N-tiles per wave ----
        {
            const int nb0 = wid * 32, nb1 = nb0 + 16;
            v8f a0 = v8zero(), a1 = v8zero();
            for (int kk = 0; kk < HH; kk += 32) {
                Frag a;
                a.q[0] = *(const u32x4*)&h_lds[nsub * HP + kk + aOff];
                a.q[1] = *(const u32x4*)&h_lds[nsub * HP + kk + 16 + aOff];
                const unsigned short* p0 = W1T + (size_t)(nb0 + nsub) * HH + kk + bOff;
                const unsigned short* p1 = W1T + (size_t)(nb1 + nsub) * HH + kk + bOff;
                Frag bA, bB;
                bA.q[0] = *(const u32x4*)p0; bA.q[1] = *(const u32x4*)(p0 + 16);
                bB.q[0] = *(const u32x4*)p1; bB.q[1] = *(const u32x4*)(p1 + 16);
                a0 = __builtin_amdgcn_wmma_f32_16x16x32_bf16(false, a.v, false, bA.v, (short)0, a0, false, false);
                a1 = __builtin_amdgcn_wmma_f32_16x16x32_bf16(false, a.v, false, bB.v, (short)0, a1, false, false);
            }
            float bb0 = b1v[nb0 + nsub], bb1 = b1v[nb1 + nsub];
#pragma unroll
            for (int v = 0; v < 8; ++v) {
                int m = mBase + v;
                z1_lds[m * Z1P + nb0 + nsub] = f2bf(tanh_fast(a0[v] + bb0));
                z1_lds[m * Z1P + nb1 + nsub] = f2bf(tanh_fast(a1[v] + bb1));
            }
        }
        __syncthreads();

        // ---- z2 = tanh(z1 @ W2 + b2) : 16x512, 1 N-tile per wave, K=1024 ----
        {
            const int nb = wid * 16;
            v8f acc = v8zero();
            for (int kk = 0; kk < H2; kk += 32) {
                Frag a;
                a.q[0] = *(const u32x4*)&z1_lds[nsub * Z1P + kk + aOff];
                a.q[1] = *(const u32x4*)&z1_lds[nsub * Z1P + kk + 16 + aOff];
                const unsigned short* p = W2T + (size_t)(nb + nsub) * H2 + kk + bOff;
                Frag bA;
                bA.q[0] = *(const u32x4*)p; bA.q[1] = *(const u32x4*)(p + 16);
                acc = __builtin_amdgcn_wmma_f32_16x16x32_bf16(false, a.v, false, bA.v, (short)0, acc, false, false);
            }
            float bb = b2v[nb + nsub];
#pragma unroll
            for (int v = 0; v < 8; ++v)
                z2_lds[(mBase + v) * HP + nb + nsub] = f2bf(tanh_fast(acc[v] + bb));
        }
        __syncthreads();

        // ---- z3 = z2 @ W3 + b3 : 16x130 (padded to 144), waves 0..8 ----
        if (wid < NPAD / 16) {
            const int nb = wid * 16;
            v8f acc = v8zero();
            for (int kk = 0; kk < HH; kk += 32) {
                Frag a;
                a.q[0] = *(const u32x4*)&z2_lds[nsub * HP + kk + aOff];
                a.q[1] = *(const u32x4*)&z2_lds[nsub * HP + kk + 16 + aOff];
                const unsigned short* p = W3T + (size_t)(nb + nsub) * HH + kk + bOff;
                Frag bA;
                bA.q[0] = *(const u32x4*)p; bA.q[1] = *(const u32x4*)(p + 16);
                acc = __builtin_amdgcn_wmma_f32_16x16x32_bf16(false, a.v, false, bA.v, (short)0, acc, false, false);
            }
            int col = nb + nsub;
            float bb = (col < NOUT) ? b3v[col] : 0.f;
#pragma unroll
            for (int v = 0; v < 8; ++v)
                z3_lds[(mBase + v) * Z3P + col] = acc[v] + bb;
        }
        __syncthreads();

        // ---- sampling: emit old x; dur = -log(u)/exp(z0); Gumbel argmax ----
        if (wid < 16) {
            const int r = wid;
            const int gb = rowBase + r;
            float bestv = -3.4e38f;
            int besti = 0;
            for (int p = lane; p < NPITCH; p += 32) {
                float gu = noise_g[((size_t)gb * GENN + s) * NPITCH + p];
                float lv = z3_lds[r * Z3P + 1 + p] - __logf(-__logf(gu));
                if (lv > bestv) { bestv = lv; besti = p; }
            }
#pragma unroll
            for (int off = 16; off > 0; off >>= 1) {
                float ov = __shfl_xor(bestv, off, 32);
                int oi = __shfl_xor(besti, off, 32);
                if (ov > bestv || (ov == bestv && oi < besti)) { bestv = ov; besti = oi; }
            }
            if (lane == 0) {
                float u = noise_u[(size_t)gb * GENN + s];
                float rate = __expf(z3_lds[r * Z3P + 0]);
                float dur = -__logf(u) / rate;
                float x0 = x_lds[r * 2 + 0];
                float x1 = x_lds[r * 2 + 1];
                float* o = out + ((size_t)gb * SEQN + HIST + s) * DD;
                o[0] = x0;               // emitted = x before update
                o[1] = x1;
                x_lds[r * 2 + 0] = dur;  // new x
                x_lds[r * 2 + 1] = (float)besti * (1.0f / 128.0f);
            }
        }
        __syncthreads();
    }
}

// -------------------------- launcher ---------------------------------------
extern "C" void kernel_launch(void* const* d_in, const int* in_sizes, int n_in,
                              void* d_out, int out_size, void* d_ws, size_t ws_size,
                              hipStream_t stream) {
    (void)in_sizes; (void)n_in; (void)out_size; (void)ws_size;
    const float* hist_x  = (const float*)d_in[0];
    const float* noise_u = (const float*)d_in[1];
    const float* noise_g = (const float*)d_in[2];
    const float* W_ih    = (const float*)d_in[3];
    const float* W_hh    = (const float*)d_in[4];
    const float* b_ih    = (const float*)d_in[5];
    const float* b_hh    = (const float*)d_in[6];
    const float* W1      = (const float*)d_in[7];
    const float* b1      = (const float*)d_in[8];
    const float* W2      = (const float*)d_in[9];
    const float* b2      = (const float*)d_in[10];
    const float* W3      = (const float*)d_in[11];
    const float* b3      = (const float*)d_in[12];
    float* out = (float*)d_out;

    char* ws = (char*)d_ws;
    size_t off = 0;
    unsigned short* Whh_bf = (unsigned short*)(ws + off); off += (size_t)G4H * HH * 2;   // 2 MB
    unsigned short* W1T    = (unsigned short*)(ws + off); off += (size_t)H2 * HH * 2;    // 1 MB
    unsigned short* W2T    = (unsigned short*)(ws + off); off += (size_t)HH * H2 * 2;    // 1 MB
    unsigned short* W3T    = (unsigned short*)(ws + off); off += (size_t)NPAD * HH * 2;  // 144 KB
    float* bsum            = (float*)(ws + off);          off += (size_t)G4H * 4;        // 8 KB

    // weight prep (bf16 convert / transpose), stream-ordered before main kernel
    {
        int n = G4H * HH;
        k_cvt_bf16<<<(n + 255) / 256, 256, 0, stream>>>(W_hh, Whh_bf, n);
    }
    {
        int n = H2 * HH;
        k_transpose_bf16<<<(n + 255) / 256, 256, 0, stream>>>(W1, W1T, H2, HH, H2, H2);
    }
    {
        int n = HH * H2;
        k_transpose_bf16<<<(n + 255) / 256, 256, 0, stream>>>(W2, W2T, HH, H2, HH, HH);
    }
    {
        int n = NPAD * HH;
        k_transpose_bf16<<<(n + 255) / 256, 256, 0, stream>>>(W3, W3T, NPAD, HH, NOUT, NOUT);
    }
    k_bias_sum<<<(G4H + 255) / 256, 256, 0, stream>>>(b_ih, b_hh, bsum, G4H);

    constexpr int HP = HH + 8;
    constexpr int Z1P = H2 + 8;
    constexpr int Z3P = 160;
    constexpr size_t SMEM_BYTES =
        (size_t)(16 * HP + 16 * Z1P + 16 * HP) * 2 + (size_t)(16 * Z3P + 32) * 4; // ~77 KB

    lstmusic_persistent<<<dim3(BB / 16), dim3(1024), SMEM_BYTES, stream>>>(
        hist_x, noise_u, noise_g, W_ih, Whh_bf, W1T, W2T, W3T, bsum, b1, b2, b3, out);
}
